// HCTransformer_36756330119667
// MI455X (gfx1250) — matrix operand order repaired and checked by reading, not compiled, not run on hardware
//
#include <hip/hip_runtime.h>

// ---------------------------------------------------------------------------
// HC-Transformer forward for MI455X (gfx1250, wave32, WMMA bf16 16x16x32,
// TDM async tensor loads feeding the GEMM LDS tiles)
// ---------------------------------------------------------------------------

typedef __bf16 bf16_t;
typedef __attribute__((ext_vector_type(16))) __bf16 v16bf;
typedef __attribute__((ext_vector_type(8)))  float  v8f;
typedef __attribute__((ext_vector_type(4)))  unsigned int u32x4;
typedef __attribute__((ext_vector_type(8)))  int i32x8;
typedef __attribute__((ext_vector_type(4)))  int i32x4;

#define LAYERS 4
#define DMODEL 1024
#define INTER  4096
#define NSTR   4
#define HEADS  16
#define HD     64
#define NC     4096          // NSTR * DMODEL
#define NOUT   24            // N*N + 2N
#define BSZ    2
#define TSEQ   1024
#define TOK    (BSZ * TSEQ)  // 2048
#define VOCAB  32000

#if __has_builtin(__builtin_amdgcn_tensor_load_to_lds) && \
    __has_builtin(__builtin_amdgcn_s_wait_tensorcnt)
#define USE_TDM 1
#else
#define USE_TDM 0
#endif

static __device__ __forceinline__ v8f wmma_bf16(v16bf a, v16bf b, v8f c) {
  return __builtin_amdgcn_wmma_f32_16x16x32_bf16(false, a, false, b, (short)0, c,
                                                 false, false);
}

// The TDM writes LDS behind the compiler's back (the LDS address is descriptor
// *data*, not a pointer). Escape the LDS base into an asm with a memory
// clobber so alias analysis treats the buffer as externally written and keeps
// the ds_load fragment reads alive and ordered after the wait/barrier.
static __device__ __forceinline__ void lds_publish(void* p) {
  asm volatile("" : : "r"(p) : "memory");
}

// A-matrix fragment (16x32 bf16), row-major source with given element stride.
// Lane L holds row (L&15); K elements {h..h+7, 16+h..16+h+7}, h = 8*(L>>4).
static __device__ __forceinline__ v16bf frag_a(const bf16_t* p, int stride, int lane) {
  union { v16bf v; uint4 q[2]; } u;
  const bf16_t* r = p + (size_t)(lane & 15) * stride + ((lane >> 4) << 3);
  u.q[0] = *(const uint4*)(r);
  u.q[1] = *(const uint4*)(r + 16);
  return u.v;
}

// B-matrix fragment (32x16 bf16), source holds columns as contiguous rows
// (i.e. B^T row-major). Lane L holds column (L&15); K elements
// {h..h+7, 16+h..16+h+7}, h = 8*(L>>4).
static __device__ __forceinline__ v16bf frag_b(const bf16_t* p, int stride, int lane) {
  union { v16bf v; uint4 q[2]; } u;
  const bf16_t* r = p + (size_t)(lane & 15) * stride + ((lane >> 4) << 3);
  u.q[0] = *(const uint4*)(r);
  u.q[1] = *(const uint4*)(r + 16);
  return u.v;
}

#if USE_TDM
// Issue one TDM descriptor: load a 128-row x 32-col bf16 tile from a row-major
// matrix (row stride K elements) into LDS at byte offset lds_off, inserting
// 4 DWORDs (8 bf16) of padding after every 16 DWORDs (32 bf16) -> LDS rows of
// 40 bf16, matching the bank-conflict-free fragment layout.
static __device__ __forceinline__ void tdm_load_tile(unsigned lds_off,
                                                     const bf16_t* gsrc, int K) {
  unsigned long long ga = (unsigned long long)gsrc;
  u32x4 g0;
  g0[0] = 1u;                                   // count=1, user descriptor
  g0[1] = lds_off;                              // lds_addr
  g0[2] = (unsigned)(ga & 0xffffffffu);         // global_addr[31:0]
  g0[3] = (unsigned)(ga >> 32) | 0x80000000u;   // global_addr[56:32] | type=2
  i32x8 g1;
  g1[0] = (1 << 16)                             // data_size = 2 bytes
        | (1 << 20)                             // pad_enable
        | (3 << 22)                             // pad_interval = 16 DWORDs
        | (3 << 25);                            // pad_amount   = 4 DWORDs
  g1[1] = (K & 0xffff) << 16;                   // tensor_dim0[15:0]
  g1[2] = ((unsigned)K >> 16) | (128 << 16);    // tensor_dim0[31:16] | tensor_dim1[15:0]
  g1[3] = (32 << 16);                           // tensor_dim1[31:16] | tile_dim0 = 32
  g1[4] = 128;                                  // tile_dim1 = 128, tile_dim2 = 0
  g1[5] = K;                                    // tensor_dim0_stride[31:0]
  g1[6] = 0;                                    // stride[47:32] | dim1_stride[15:0]
  g1[7] = 0;
  i32x4 z4 = {0, 0, 0, 0};
  i32x8 z8 = {0, 0, 0, 0, 0, 0, 0, 0};
  __builtin_amdgcn_tensor_load_to_lds(g0, g1, z4, z4, z8, 0);
}
#endif

// ---------------------------------------------------------------------------
// Embedding gather, broadcast to N streams
// ---------------------------------------------------------------------------
__global__ __launch_bounds__(256) void k_embed(const int* __restrict__ ids,
                                               const float* __restrict__ emb,
                                               float* __restrict__ xs) {
  int gid = blockIdx.x * 256 + threadIdx.x;        // TOK*DMODEL threads
  int tok = gid >> 10, c = gid & 1023;
  float v = emb[(size_t)ids[tok] * DMODEL + c];
#pragma unroll
  for (int n = 0; n < NSTR; ++n)
    xs[((size_t)tok * NSTR + n) * DMODEL + c] = v;
}

// ---------------------------------------------------------------------------
// Weight transpose + f32 -> bf16:  W[K,N] row-major -> Wt[N,K] row-major
// ---------------------------------------------------------------------------
__global__ __launch_bounds__(256) void k_wt(const float* __restrict__ W,
                                            bf16_t* __restrict__ Wt,
                                            int K, int N) {
  __shared__ float tile[32][33];
  int kb = blockIdx.y * 32, nb = blockIdx.x * 32;
  int tx = threadIdx.x & 31, ty = threadIdx.x >> 5;   // ty in 0..7
#pragma unroll
  for (int i = 0; i < 32; i += 8)
    tile[ty + i][tx] = W[(size_t)(kb + ty + i) * N + nb + tx];
  __syncthreads();
#pragma unroll
  for (int i = 0; i < 32; i += 8)
    Wt[(size_t)(nb + ty + i) * K + kb + tx] = (bf16_t)tile[tx][ty + i];
}

// ---------------------------------------------------------------------------
// mHC maps: RMS(4096) -> xp @ phi(4096x24) -> gates + Sinkhorn(4x4, 20 iters)
// One block of 256 threads per token.
// ---------------------------------------------------------------------------
__global__ __launch_bounds__(256) void k_mhc(const float* __restrict__ xs,
                                             const float* __restrict__ wrms,
                                             const float* __restrict__ phi,
                                             const float* __restrict__ bb,
                                             const float* __restrict__ alpha,
                                             float* __restrict__ Hpre,
                                             float* __restrict__ Hpost,
                                             float* __restrict__ Hres) {
  int tok = blockIdx.x;
  int tid = threadIdx.x, lane = tid & 31, wv = tid >> 5;
  const float* x = xs + (size_t)tok * NC;

  __shared__ float red[8];
  __shared__ float ttacc[8][NOUT];
  __shared__ float tt[NOUT];

  float ss = 0.f;
  for (int i = tid; i < NC; i += 256) { float v = x[i]; ss += v * v; }
#pragma unroll
  for (int o = 16; o > 0; o >>= 1) ss += __shfl_xor(ss, o);
  if (lane == 0) red[wv] = ss;
  __syncthreads();
  if (tid == 0) {
    float t = 0.f;
    for (int w = 0; w < 8; ++w) t += red[w];
    red[0] = rsqrtf(t * (1.f / NC) + 1e-6f);
  }
  __syncthreads();
  float rstd = red[0];

  float acc[NOUT];
#pragma unroll
  for (int j = 0; j < NOUT; ++j) acc[j] = 0.f;
  for (int i = tid; i < NC; i += 256) {
    float xp = x[i] * rstd * wrms[i];
    const float* ph = phi + (size_t)i * NOUT;
#pragma unroll
    for (int j = 0; j < NOUT; ++j) acc[j] += xp * ph[j];
  }
#pragma unroll
  for (int j = 0; j < NOUT; ++j) {
    float v = acc[j];
#pragma unroll
    for (int o = 16; o > 0; o >>= 1) v += __shfl_xor(v, o);
    if (lane == 0) ttacc[wv][j] = v;
  }
  __syncthreads();
  if (tid < NOUT) {
    float t = 0.f;
    for (int w = 0; w < 8; ++w) t += ttacc[w][tid];
    tt[tid] = t;
  }
  __syncthreads();

  if (tid == 0) {
    float a0 = alpha[0], a1 = alpha[1], a2 = alpha[2];
#pragma unroll
    for (int n = 0; n < NSTR; ++n) {
      Hpre[tok * NSTR + n]  = 1.f / (1.f + expf(-(a0 * tt[n] + bb[n])));
      Hpost[tok * NSTR + n] = 2.f / (1.f + expf(-(a1 * tt[NSTR + n] + bb[NSTR + n])));
    }
    float M[16], mx = -1e30f;
#pragma unroll
    for (int i = 0; i < 16; ++i) {
      float v = a2 * tt[8 + i] + bb[8 + i];
      M[i] = v; mx = fmaxf(mx, v);
    }
#pragma unroll
    for (int i = 0; i < 16; ++i) M[i] = expf(M[i] - mx);
    for (int it = 0; it < 20; ++it) {
#pragma unroll
      for (int r = 0; r < 4; ++r) {
        float s = M[r*4] + M[r*4+1] + M[r*4+2] + M[r*4+3];
        float inv = 1.f / fmaxf(s, 1e-12f);
        M[r*4] *= inv; M[r*4+1] *= inv; M[r*4+2] *= inv; M[r*4+3] *= inv;
      }
#pragma unroll
      for (int c = 0; c < 4; ++c) {
        float s = M[c] + M[4+c] + M[8+c] + M[12+c];
        float inv = 1.f / fmaxf(s, 1e-12f);
        M[c] *= inv; M[4+c] *= inv; M[8+c] *= inv; M[12+c] *= inv;
      }
    }
#pragma unroll
    for (int i = 0; i < 16; ++i) Hres[tok * 16 + i] = M[i];
  }
}

// ---------------------------------------------------------------------------
// x_in = sum_n Hpre[n]*xs[n]; RMS over D; write bf16 (x_in * rstd * w_norm)
// One block per token.
// ---------------------------------------------------------------------------
__global__ __launch_bounds__(256) void k_pre(const float* __restrict__ xs,
                                             const float* __restrict__ Hpre,
                                             const float* __restrict__ wnorm,
                                             bf16_t* __restrict__ xhat) {
  int tok = blockIdx.x, tid = threadIdx.x, lane = tid & 31, wv = tid >> 5;
  const float* x = xs + (size_t)tok * NC;
  const float* hp = Hpre + tok * NSTR;
  float h0 = hp[0], h1 = hp[1], h2 = hp[2], h3 = hp[3];

  float v[4]; float ss = 0.f;
#pragma unroll
  for (int u = 0; u < 4; ++u) {
    int c = tid + u * 256;
    float t = h0 * x[c] + h1 * x[DMODEL + c] + h2 * x[2*DMODEL + c] + h3 * x[3*DMODEL + c];
    v[u] = t; ss += t * t;
  }
  __shared__ float red[8];
#pragma unroll
  for (int o = 16; o > 0; o >>= 1) ss += __shfl_xor(ss, o);
  if (lane == 0) red[wv] = ss;
  __syncthreads();
  if (tid == 0) {
    float t = 0.f;
    for (int w = 0; w < 8; ++w) t += red[w];
    red[0] = rsqrtf(t * (1.f / DMODEL) + 1e-6f);
  }
  __syncthreads();
  float rstd = red[0];
#pragma unroll
  for (int u = 0; u < 4; ++u) {
    int c = tid + u * 256;
    xhat[(size_t)tok * DMODEL + c] = (bf16_t)(v[u] * rstd * wnorm[c]);
  }
}

// ---------------------------------------------------------------------------
// GEMM: C[M,N] f32 = A[M,K] bf16 (row-major) x Bt[N,K] bf16 (B transposed).
// Block tile 128x128, 256 threads = 8 waves as 4(M) x 2(N); each wave owns a
// 32x64 output tile: 8 accumulators, 8 wmma per 32-K step. LDS double
// buffered; tiles are staged by the Tensor Data Mover (wave 0 issues two
// descriptors per K step, s_wait_tensorcnt before the block barrier), with a
// VGPR-staged fallback when the TDM builtin is unavailable.
// ---------------------------------------------------------------------------
__global__ __launch_bounds__(256) void k_gemm(const bf16_t* __restrict__ A,
                                              const bf16_t* __restrict__ Bt,
                                              float* __restrict__ C,
                                              int M, int Nn, int K) {
  // [buf][mat][row][col]; one 128x40 bf16 plane = 10240 B.
  __shared__ __align__(16) bf16_t smem[2][2][128][40];
  const unsigned PLANE = 128 * 40 * 2;          // bytes
  int bm = blockIdx.y * 128, bn = blockIdx.x * 128;
  int tid = threadIdx.x, lane = tid & 31, wave = tid >> 5;
  int wm = wave & 3, wn = wave >> 2;

  const bf16_t* gA = A  + (size_t)bm * K;
  const bf16_t* gB = Bt + (size_t)bn * K;

  v8f acc[2][4];
#pragma unroll
  for (int a = 0; a < 2; ++a)
#pragma unroll
    for (int j = 0; j < 4; ++j) acc[a][j] = v8f{};

#if USE_TDM
  if (wave == 0) {
    tdm_load_tile(0 * PLANE, gA, K);
    tdm_load_tile(1 * PLANE, gB, K);
    __builtin_amdgcn_s_wait_tensorcnt(0);
  }
  __syncthreads();
#else
  int sr = tid >> 1;              // staging row 0..127
  int sc = (tid & 1) << 4;        // staging col 0 or 16
  {
    const uint4* s0 = (const uint4*)(gA + (size_t)sr * K + sc);
    const uint4* s1 = (const uint4*)(gB + (size_t)sr * K + sc);
    uint4 t0 = s0[0], t1 = s0[1], t2 = s1[0], t3 = s1[1];
    uint4* d0 = (uint4*)&smem[0][0][sr][sc];
    uint4* d1 = (uint4*)&smem[0][1][sr][sc];
    d0[0] = t0; d0[1] = t1; d1[0] = t2; d1[1] = t3;
  }
  __syncthreads();
#endif

  int buf = 0;
  for (int k0 = 0; k0 < K; k0 += 32) {
    lds_publish(&smem[0][0][0][0]);   // DMA-written LDS: keep ds_loads alive
    // stage next K step into the other buffer while computing this one
    if (k0 + 32 < K) {
#if USE_TDM
      if (wave == 0) {
        tdm_load_tile((unsigned)((buf ^ 1) * 2 + 0) * PLANE, gA + k0 + 32, K);
        tdm_load_tile((unsigned)((buf ^ 1) * 2 + 1) * PLANE, gB + k0 + 32, K);
      }
#else
      const uint4* s0 = (const uint4*)(gA + (size_t)sr * K + k0 + 32 + sc);
      const uint4* s1 = (const uint4*)(gB + (size_t)sr * K + k0 + 32 + sc);
      uint4 t0 = s0[0], t1 = s0[1], t2 = s1[0], t3 = s1[1];
      uint4* d0 = (uint4*)&smem[buf ^ 1][0][sr][sc];
      uint4* d1 = (uint4*)&smem[buf ^ 1][1][sr][sc];
      d0[0] = t0; d0[1] = t1; d1[0] = t2; d1[1] = t3;
#endif
      if (k0 + 64 < K) {                        // speculative prefetch K+64
        __builtin_prefetch(gA + (size_t)(tid >> 1) * K + k0 + 64, 0, 1);
        __builtin_prefetch(gB + (size_t)(tid >> 1) * K + k0 + 64, 0, 1);
      }
    }

    v16bf af0 = frag_a(&smem[buf][0][wm * 32][0], 40, lane);
    v16bf af1 = frag_a(&smem[buf][0][wm * 32 + 16][0], 40, lane);
#pragma unroll
    for (int j = 0; j < 4; ++j) {
      v16bf bfm = frag_b(&smem[buf][1][wn * 64 + j * 16][0], 40, lane);
      acc[0][j] = wmma_bf16(af0, bfm, acc[0][j]);
      acc[1][j] = wmma_bf16(af1, bfm, acc[1][j]);
    }
#if USE_TDM
    if (wave == 0) __builtin_amdgcn_s_wait_tensorcnt(0);
#endif
    __syncthreads();
    buf ^= 1;
  }

#pragma unroll
  for (int a = 0; a < 2; ++a) {
    int mbase = bm + wm * 32 + a * 16 + ((lane >> 4) << 3);
    int nbase = bn + wn * 64 + (lane & 15);
#pragma unroll
    for (int j = 0; j < 4; ++j)
#pragma unroll
      for (int r = 0; r < 8; ++r)
        C[(size_t)(mbase + r) * Nn + nbase + j * 16] = acc[a][j][r];
  }
}

// ---------------------------------------------------------------------------
// Split QKV f32 [tok,3D] -> q bf16 [B,H,T,hd] (pre-scaled 1/sqrt(hd)),
// k bf16 [B,H,T,hd], v^T bf16 [B,H,hd,T]
// ---------------------------------------------------------------------------
__global__ __launch_bounds__(256) void k_split(const float* __restrict__ qkv,
                                               bf16_t* __restrict__ q,
                                               bf16_t* __restrict__ k,
                                               bf16_t* __restrict__ vt) {
  int gid = blockIdx.x * 256 + threadIdx.x;   // TOK*DMODEL
  int tok = gid >> 10, c = gid & 1023;
  int h = c >> 6, dd = c & 63;
  int b = tok >> 10, t = tok & 1023;
  size_t src = (size_t)tok * (3 * DMODEL);
  size_t bh  = (size_t)(b * HEADS + h);
  q[(bh * TSEQ + t) * HD + dd]  = (bf16_t)(qkv[src + c] * 0.125f);
  k[(bh * TSEQ + t) * HD + dd]  = (bf16_t)(qkv[src + DMODEL + c]);
  vt[(bh * HD + dd) * TSEQ + t] = (bf16_t)(qkv[src + 2 * DMODEL + c]);
}

// ---------------------------------------------------------------------------
// Flash attention (causal). One wave per 16-row q-tile; 4 waves per block.
// QK^T and PV via WMMA bf16; online softmax via cross-lane shuffles;
// P relayout through a 1KB per-wave LDS tile. Output bf16 into xhat.
// ---------------------------------------------------------------------------
__global__ __launch_bounds__(128) void k_flash(const bf16_t* __restrict__ q,
                                               const bf16_t* __restrict__ k,
                                               const bf16_t* __restrict__ vt,
                                               bf16_t* __restrict__ out) {
  __shared__ __align__(16) bf16_t Ps[4][16][40];
  int lane = threadIdx.x & 31, wv = threadIdx.x >> 5;
  int qt = blockIdx.x * 4 + wv;                 // 0 .. B*H*(T/16)-1
  int qi = qt & 63;
  int h  = (qt >> 6) & (HEADS - 1);
  int b  = qt >> 10;

  const bf16_t* qp = q  + (((size_t)(b * HEADS + h)) * TSEQ + qi * 16) * HD;
  const bf16_t* kp = k  + ((size_t)(b * HEADS + h)) * TSEQ * HD;
  const bf16_t* vp = vt + ((size_t)(b * HEADS + h)) * HD * TSEQ;

  v16bf qf0 = frag_a(qp, HD, lane);
  v16bf qf1 = frag_a(qp + 32, HD, lane);

  float mrun[8], lrun[8];
  v8f acc[4] = {v8f{}, v8f{}, v8f{}, v8f{}};
#pragma unroll
  for (int r = 0; r < 8; ++r) { mrun[r] = -1e30f; lrun[r] = 0.f; }

  int hiRow = (lane >> 4) << 3;       // 0 or 8
  int ncol  = lane & 15;
  int nkt = (qi * 16 + 47) >> 5;      // #32-key chunks touching causal band

  for (int kt = 0; kt < nkt; ++kt) {
    int kbase = kt * 32;
    v8f S0 = v8f{}, S1 = v8f{};
    {
      v16bf kf;
      kf = frag_b(kp + (size_t)kbase * HD,            HD, lane);
      S0 = wmma_bf16(qf0, kf, S0);
      kf = frag_b(kp + (size_t)kbase * HD + 32,       HD, lane);
      S0 = wmma_bf16(qf1, kf, S0);
      kf = frag_b(kp + (size_t)(kbase + 16) * HD,     HD, lane);
      S1 = wmma_bf16(qf0, kf, S1);
      kf = frag_b(kp + (size_t)(kbase + 16) * HD + 32, HD, lane);
      S1 = wmma_bf16(qf1, kf, S1);
    }
#pragma unroll
    for (int r = 0; r < 8; ++r) {
      int qrow = qi * 16 + hiRow + r;
      float s0 = S0[r]; if (kbase + ncol > qrow)      s0 = -1e30f;
      float s1 = S1[r]; if (kbase + 16 + ncol > qrow) s1 = -1e30f;
      float mx = fmaxf(s0, s1);
      mx = fmaxf(mx, __shfl_xor(mx, 1));
      mx = fmaxf(mx, __shfl_xor(mx, 2));
      mx = fmaxf(mx, __shfl_xor(mx, 4));
      mx = fmaxf(mx, __shfl_xor(mx, 8));
      float mnew = fmaxf(mrun[r], mx);
      float p0 = __expf(s0 - mnew), p1 = __expf(s1 - mnew);
      float ls = p0 + p1;
      ls += __shfl_xor(ls, 1);
      ls += __shfl_xor(ls, 2);
      ls += __shfl_xor(ls, 4);
      ls += __shfl_xor(ls, 8);
      float corr = __expf(mrun[r] - mnew);
      lrun[r] = lrun[r] * corr + ls;
      mrun[r] = mnew;
#pragma unroll
      for (int j = 0; j < 4; ++j) acc[j][r] *= corr;
      Ps[wv][hiRow + r][ncol]      = (bf16_t)p0;
      Ps[wv][hiRow + r][16 + ncol] = (bf16_t)p1;
    }
    asm volatile("s_wait_dscnt 0" ::: "memory");   // wave-local LDS RAW fence
    v16bf pf = frag_a(&Ps[wv][0][0], 40, lane);
#pragma unroll
    for (int j = 0; j < 4; ++j) {
      v16bf vf = frag_b(vp + (size_t)(j * 16) * TSEQ + kbase, TSEQ, lane);
      acc[j] = wmma_bf16(pf, vf, acc[j]);
    }
  }

#pragma unroll
  for (int j = 0; j < 4; ++j)
#pragma unroll
    for (int r = 0; r < 8; ++r) {
      int t = qi * 16 + hiRow + r;
      int c = h * HD + j * 16 + ncol;
      out[((size_t)(b * TSEQ + t)) * DMODEL + c] = (bf16_t)(acc[j][r] / lrun[r]);
    }
}

// ---------------------------------------------------------------------------
// SiLU + f32 -> bf16
// ---------------------------------------------------------------------------
__global__ __launch_bounds__(256) void k_silu(const float* __restrict__ X,
                                              bf16_t* __restrict__ Y) {
  int gid = blockIdx.x * 256 + threadIdx.x;
  float x = X[gid];
  Y[gid] = (bf16_t)(x / (1.f + __expf(-x)));
}

// ---------------------------------------------------------------------------
// Stream remix: xs_next[i] = sum_j Hres[i,j]*xs[j] + y*Hpost[i]
// ---------------------------------------------------------------------------
__global__ __launch_bounds__(256) void k_combine(const float* __restrict__ xs,
                                                 const float* __restrict__ Hres,
                                                 const float* __restrict__ Hpost,
                                                 const float* __restrict__ y,
                                                 float* __restrict__ xo) {
  int gid = blockIdx.x * 256 + threadIdx.x;    // TOK*DMODEL
  int tok = gid >> 10, c = gid & 1023;
  const float* hr = Hres + tok * 16;
  const float* hp = Hpost + tok * NSTR;
  float yv = y[gid];
  float x0 = xs[((size_t)tok * NSTR + 0) * DMODEL + c];
  float x1 = xs[((size_t)tok * NSTR + 1) * DMODEL + c];
  float x2 = xs[((size_t)tok * NSTR + 2) * DMODEL + c];
  float x3 = xs[((size_t)tok * NSTR + 3) * DMODEL + c];
#pragma unroll
  for (int i = 0; i < NSTR; ++i)
    xo[((size_t)tok * NSTR + i) * DMODEL + c] =
        hr[i * 4 + 0] * x0 + hr[i * 4 + 1] * x1 +
        hr[i * 4 + 2] * x2 + hr[i * 4 + 3] * x3 + yv * hp[i];
}

// ---------------------------------------------------------------------------
// Mean over streams, f32 -> bf16
// ---------------------------------------------------------------------------
__global__ __launch_bounds__(256) void k_mean(const float* __restrict__ xs,
                                              bf16_t* __restrict__ xhat) {
  int gid = blockIdx.x * 256 + threadIdx.x;    // TOK*DMODEL
  int tok = gid >> 10, c = gid & 1023;
  float s = xs[((size_t)tok * NSTR + 0) * DMODEL + c] +
            xs[((size_t)tok * NSTR + 1) * DMODEL + c] +
            xs[((size_t)tok * NSTR + 2) * DMODEL + c] +
            xs[((size_t)tok * NSTR + 3) * DMODEL + c];
  xhat[gid] = (bf16_t)(s * 0.25f);
}

// ---------------------------------------------------------------------------
// Host orchestration
// ---------------------------------------------------------------------------
extern "C" void kernel_launch(void* const* d_in, const int* in_sizes, int n_in,
                              void* d_out, int out_size, void* d_ws, size_t ws_size,
                              hipStream_t stream) {
  (void)in_sizes; (void)n_in; (void)out_size; (void)ws_size;

  const int*   ids        = (const int*)  d_in[0];
  const float* embed      = (const float*)d_in[1];
  const float* unembed    = (const float*)d_in[2];
  const float* attn_rms   = (const float*)d_in[3];
  const float* attn_phi   = (const float*)d_in[4];
  const float* attn_bb    = (const float*)d_in[5];
  const float* attn_alpha = (const float*)d_in[6];
  const float* ffn_rms    = (const float*)d_in[7];
  const float* ffn_phi    = (const float*)d_in[8];
  const float* ffn_bb     = (const float*)d_in[9];
  const float* ffn_alpha  = (const float*)d_in[10];
  const float* attn_norm  = (const float*)d_in[11];
  const float* attn_qkv   = (const float*)d_in[12];
  const float* attn_out   = (const float*)d_in[13];
  const float* ffn_norm   = (const float*)d_in[14];
  const float* ffn_fc1    = (const float*)d_in[15];
  const float* ffn_fc2    = (const float*)d_in[16];

  char* ws = (char*)d_ws;
  size_t off = 0;
  auto carve = [&](size_t bytes) -> char* {
    char* p = ws + off;
    off += (bytes + 255) & ~(size_t)255;
    return p;
  };
  float*  xs0   = (float*) carve((size_t)TOK * NC * 4);          // 32 MB
  float*  xs1   = (float*) carve((size_t)TOK * NC * 4);          // 32 MB
  float*  Hpre  = (float*) carve((size_t)TOK * NSTR * 4);
  float*  Hpost = (float*) carve((size_t)TOK * NSTR * 4);
  float*  Hres  = (float*) carve((size_t)TOK * 16 * 4);
  bf16_t* xhat  = (bf16_t*)carve((size_t)TOK * INTER * 2);       // 16 MB (A input)
  float*  Cbuf  = (float*) carve((size_t)TOK * INTER * 4);       // 32 MB
  float*  ybuf  = (float*) carve((size_t)TOK * DMODEL * 4);      // 8 MB
  bf16_t* qb    = (bf16_t*)carve((size_t)BSZ * HEADS * TSEQ * HD * 2);
  bf16_t* kb    = (bf16_t*)carve((size_t)BSZ * HEADS * TSEQ * HD * 2);
  bf16_t* vtb   = (bf16_t*)carve((size_t)BSZ * HEADS * HD * TSEQ * 2);
  bf16_t* Wt    = (bf16_t*)carve((size_t)VOCAB * DMODEL * 2);    // 62.5 MB, reused

  const int EW = TOK * DMODEL / 256;   // elementwise grid over [TOK, D]

  auto cvt = [&](const float* W, int K, int N) {
    k_wt<<<dim3(N / 32, K / 32), 256, 0, stream>>>(W, Wt, K, N);
  };
  auto gemm = [&](const bf16_t* A, float* C, int M, int N, int K) {
    k_gemm<<<dim3(N / 128, M / 128), 256, 0, stream>>>(A, Wt, C, M, N, K);
  };

  float* cur = xs0;
  float* nxt = xs1;
  k_embed<<<EW, 256, 0, stream>>>(ids, embed, cur);

  for (int l = 0; l < LAYERS; ++l) {
    // ---- attention hblock ----
    k_mhc<<<TOK, 256, 0, stream>>>(cur, attn_rms + (size_t)l * NC,
                                   attn_phi + (size_t)l * NC * NOUT,
                                   attn_bb + (size_t)l * NOUT,
                                   attn_alpha + (size_t)l * 3,
                                   Hpre, Hpost, Hres);
    k_pre<<<TOK, 256, 0, stream>>>(cur, Hpre, attn_norm + (size_t)l * DMODEL, xhat);
    cvt(attn_qkv + (size_t)l * DMODEL * 3 * DMODEL, DMODEL, 3 * DMODEL);
    gemm(xhat, Cbuf, TOK, 3 * DMODEL, DMODEL);
    k_split<<<EW, 256, 0, stream>>>(Cbuf, qb, kb, vtb);
    k_flash<<<(BSZ * HEADS * (TSEQ / 16)) / 4, 128, 0, stream>>>(qb, kb, vtb, xhat);
    cvt(attn_out + (size_t)l * DMODEL * DMODEL, DMODEL, DMODEL);
    gemm(xhat, ybuf, TOK, DMODEL, DMODEL);
    k_combine<<<EW, 256, 0, stream>>>(cur, Hres, Hpost, ybuf, nxt);
    { float* t = cur; cur = nxt; nxt = t; }

    // ---- FFN hblock ----
    k_mhc<<<TOK, 256, 0, stream>>>(cur, ffn_rms + (size_t)l * NC,
                                   ffn_phi + (size_t)l * NC * NOUT,
                                   ffn_bb + (size_t)l * NOUT,
                                   ffn_alpha + (size_t)l * 3,
                                   Hpre, Hpost, Hres);
    k_pre<<<TOK, 256, 0, stream>>>(cur, Hpre, ffn_norm + (size_t)l * DMODEL, xhat);
    cvt(ffn_fc1 + (size_t)l * DMODEL * INTER, DMODEL, INTER);
    gemm(xhat, Cbuf, TOK, INTER, DMODEL);
    k_silu<<<TOK * INTER / 256, 256, 0, stream>>>(Cbuf, xhat);
    cvt(ffn_fc2 + (size_t)l * INTER * DMODEL, INTER, DMODEL);
    gemm(xhat, ybuf, TOK, DMODEL, INTER);
    k_combine<<<EW, 256, 0, stream>>>(cur, Hres, Hpost, ybuf, nxt);
    { float* t = cur; cur = nxt; nxt = t; }
  }

  // ---- mean over streams + unembed ----
  k_mean<<<EW, 256, 0, stream>>>(cur, xhat);
  cvt(unembed, DMODEL, VOCAB);
  gemm(xhat, (float*)d_out, TOK, VOCAB, DMODEL);
}
